// BotRGCN1_32495722562031
// MI455X (gfx1250) — matrix operand order repaired and checked by reading, not compiled, not run on hardware
//
#include <hip/hip_runtime.h>

#define LRELU_SLOPE 0.01f
#define DMODEL 128
#define NREL 5

typedef __attribute__((ext_vector_type(16))) __bf16 v16bf;
typedef __attribute__((ext_vector_type(8)))  __bf16 v8bf;
typedef __attribute__((ext_vector_type(4)))  __bf16 v4bf;
typedef __attribute__((ext_vector_type(8)))  float  v8f;

__device__ __forceinline__ float leaky_f(float x) { return x >= 0.f ? x : LRELU_SLOPE * x; }

// ---------------------------------------------------------------------------
// Pack a [K,128] f32 weight matrix into per-lane WMMA B fragments:
//   Bp[((nt*(K/32) + ks)*32 + lane)*16 + e] = bf16( B[(ks*32 + (lane>>4)*16 + e)*128 + nt*16 + (lane&15)] )
// so the GEMM inner loop reads one contiguous 32-byte v16bf per lane.
// ---------------------------------------------------------------------------
__global__ void pack_b(const float* __restrict__ B, __bf16* __restrict__ Bp, int K)
{
    const int nks = K >> 5;
    const int total = 8 * nks * 32 * 16;
    int tid = blockIdx.x * blockDim.x + threadIdx.x;
    if (tid >= total) return;
    const int e    = tid & 15;
    const int lane = (tid >> 4) & 31;
    const int f    = tid >> 9;          // nt*nks + ks
    const int ks   = f % nks;
    const int nt   = f / nks;
    const int k    = (ks << 5) + ((lane >> 4) << 4) + e;
    const int col  = (nt << 4) + (lane & 15);
    Bp[tid] = (__bf16)B[(size_t)k * DMODEL + col];
}

// ---------------------------------------------------------------------------
// C[M,128] = act(A[M,K] @ B[K,128] + bias), B pre-packed by pack_b.
// Block = 256 threads = 8 waves; block tile 32(M) x 128(N); wave tile 32x16
// (two WMMAs sharing one B fragment). A staged via double-buffered LDS (bf16).
// ---------------------------------------------------------------------------
template<int ACT, int HASB>
__global__ __launch_bounds__(256)
void gemm_nc128(const float* __restrict__ A, const __bf16* __restrict__ Bp,
                const float* __restrict__ bias, float* __restrict__ Cmat,
                int M, int K)
{
    __shared__ __bf16 smA[2][32 * 32];      // [buf][row*32 + k]

    const int m0   = blockIdx.x * 32;
    const int wv   = threadIdx.x >> 5;      // 0..7 -> 16-col tile
    const int lane = threadIdx.x & 31;
    const int lx   = lane & 15;
    const int kh   = lane >> 4;
    const int nks  = K >> 5;

    // cooperative A staging: thread -> (row 0..31, 4 consecutive k)
    const int srow = threadIdx.x >> 3;
    const int skc  = (threadIdx.x & 7) << 2;
    int garow = m0 + srow;
    if (garow >= M) garow = M - 1;          // clamp keeps all lanes live (WMMA needs EXEC all-1s)
    const float* __restrict__ Astage = A + (size_t)garow * K + skc;

    const v16bf* __restrict__ Bfrag = (const v16bf*)Bp + (size_t)wv * nks * 32 + lane;

    v8f acc0 = {}, acc1 = {};
    for (int ks = 0; ks < nks; ++ks) {
        // stage 32x32 f32 -> bf16 into LDS (coalesced b128 global load, b64 ds store)
        const float4 a4 = *(const float4*)(Astage + ((size_t)ks << 5));
        v4bf pk = { (__bf16)a4.x, (__bf16)a4.y, (__bf16)a4.z, (__bf16)a4.w };
        *(v4bf*)&smA[ks & 1][srow * 32 + skc] = pk;
        __syncthreads();

        // B fragment: one contiguous 32B load from packed weights (L2-resident)
        const v16bf bf = Bfrag[(size_t)ks * 32];

        // A fragments: two contiguous 16B LDS reads per 16-row tile
        const __bf16* sa = smA[ks & 1];
        const v8bf a0lo = *(const v8bf*)(sa + lx * 32 + kh * 16);
        const v8bf a0hi = *(const v8bf*)(sa + lx * 32 + 8 + kh * 16);
        const v8bf a1lo = *(const v8bf*)(sa + (16 + lx) * 32 + kh * 16);
        const v8bf a1hi = *(const v8bf*)(sa + (16 + lx) * 32 + 8 + kh * 16);
        const v16bf af0 = __builtin_shufflevector(a0lo, a0hi, 0,1,2,3,4,5,6,7,8,9,10,11,12,13,14,15);
        const v16bf af1 = __builtin_shufflevector(a1lo, a1hi, 0,1,2,3,4,5,6,7,8,9,10,11,12,13,14,15);

        acc0 = __builtin_amdgcn_wmma_f32_16x16x32_bf16(false, af0, false, bf, (short)0, acc0, false, false);
        acc1 = __builtin_amdgcn_wmma_f32_16x16x32_bf16(false, af1, false, bf, (short)0, acc1, false, false);
    }

    const int col = (wv << 4) + lx;
    float bv = 0.f;
    if (HASB) bv = bias[col];

    float* __restrict__ Cp = Cmat + (size_t)(m0 + (kh << 3)) * DMODEL + col;
    if (m0 + 32 <= M) {
#pragma unroll
        for (int i = 0; i < 8; ++i) {
            float v0 = acc0[i] + bv;
            float v1 = acc1[i] + bv;
            if (ACT) { v0 = leaky_f(v0); v1 = leaky_f(v1); }
            Cp[(size_t)i * DMODEL]            = v0;
            Cp[(size_t)(i + 16) * DMODEL]     = v1;
        }
    } else {
#pragma unroll
        for (int i = 0; i < 8; ++i) {
            const int r0 = m0 + (kh << 3) + i;
            float v0 = acc0[i] + bv;
            float v1 = acc1[i] + bv;
            if (ACT) { v0 = leaky_f(v0); v1 = leaky_f(v1); }
            if (r0 < M)      Cmat[(size_t)r0 * DMODEL + col]        = v0;
            if (r0 + 16 < M) Cmat[(size_t)(r0 + 16) * DMODEL + col] = v1;
        }
    }
}

__global__ void zerof(float* __restrict__ p, int n)
{
    int i = blockIdx.x * blockDim.x + threadIdx.x;
    if (i < n) p[i] = 0.f;
}

__global__ void count_edges(const int* __restrict__ dst, const int* __restrict__ et,
                            float* __restrict__ cnt, int E)
{
    int e = blockIdx.x * blockDim.x + threadIdx.x;
    if (e < E) atomicAdd(&cnt[(size_t)dst[e] * NREL + et[e]], 1.0f);
}

// One wave per edge: out[dst] += H[src] / max(cnt[dst,r],1) for edges of type r.
__global__ __launch_bounds__(256)
void scatter_rel(const int* __restrict__ src, const int* __restrict__ dst,
                 const int* __restrict__ et, const float* __restrict__ cnt,
                 const float* __restrict__ H, float* __restrict__ out,
                 int E, int r)
{
    const int wid  = (int)((blockIdx.x * blockDim.x + threadIdx.x) >> 5);
    const int lane = threadIdx.x & 31;
    if (wid >= E) return;
    if (et[wid] != r) return;                 // uniform per wave
    const int s = src[wid], d = dst[wid];
    const float inv = 1.0f / fmaxf(cnt[(size_t)d * NREL + r], 1.0f);
    const float4 h = ((const float4*)(H + (size_t)s * DMODEL))[lane];
    float* o = out + (size_t)d * DMODEL + lane * 4;
    atomicAdd(o + 0, h.x * inv);
    atomicAdd(o + 1, h.y * inv);
    atomicAdd(o + 2, h.z * inv);
    atomicAdd(o + 3, h.w * inv);
}

// out[N,2] = X[N,128] @ W[128,2] + b
__global__ void out_proj2(const float* __restrict__ X, const float* __restrict__ W,
                          const float* __restrict__ b, float* __restrict__ out, int M)
{
    int n = blockIdx.x * blockDim.x + threadIdx.x;
    if (n >= M) return;
    float a0 = b[0], a1 = b[1];
    const float* x = X + (size_t)n * DMODEL;
#pragma unroll 8
    for (int k = 0; k < DMODEL; ++k) {
        const float v = x[k];
        a0 += v * W[2 * k];
        a1 += v * W[2 * k + 1];
    }
    out[2 * n]     = a0;
    out[2 * n + 1] = a1;
}

extern "C" void kernel_launch(void* const* d_in, const int* in_sizes, int n_in,
                              void* d_out, int out_size, void* d_ws, size_t ws_size,
                              hipStream_t stream)
{
    (void)n_in; (void)out_size; (void)ws_size;

    const float* des    = (const float*)d_in[0];
    // d_in[1..3] (tweet, num_prop, cat_prop) unused by the reference graph
    const int*   eidx   = (const int*)d_in[4];     // [2,E]
    const int*   etyp   = (const int*)d_in[5];     // [E]
    const float* W_des  = (const float*)d_in[6];
    const float* b_des  = (const float*)d_in[7];
    const float* W_in   = (const float*)d_in[8];
    const float* b_in   = (const float*)d_in[9];
    const float* W_rel  = (const float*)d_in[10];  // [R,128,128]
    const float* W_root = (const float*)d_in[11];
    const float* b_rgcn = (const float*)d_in[12];
    const float* W_out1 = (const float*)d_in[13];
    const float* b_out1 = (const float*)d_in[14];
    const float* W_out2 = (const float*)d_in[15];  // [128,2]
    const float* b_out2 = (const float*)d_in[16];

    const int DDES = 768;
    const int N = in_sizes[0] / DDES;
    const int E = in_sizes[5];
    const int* srcp = eidx;
    const int* dstp = eidx + E;

    float* bufA = (float*)d_ws;                    // [N,128]
    float* bufB = bufA + (size_t)N * DMODEL;       // [N,128]
    float* Hbuf = bufB + (size_t)N * DMODEL;       // [N,128] (per-relation reuse)
    float* cnt  = Hbuf + (size_t)N * DMODEL;       // [N,5]

    // packed bf16 weight region (64B aligned)
    size_t off = ((size_t)(cnt + (size_t)N * NREL - (float*)d_ws) * 4 + 63) & ~(size_t)63;
    __bf16* packBase = (__bf16*)((char*)d_ws + off);
    const size_t PK768 = 8 * (768 / 32) * 32 * 16;   // 98304 elems
    const size_t PK128 = 8 * (128 / 32) * 32 * 16;   // 16384 elems
    __bf16* Bp_des  = packBase;
    __bf16* Bp_in   = Bp_des  + PK768;
    __bf16* Bp_root = Bp_in   + PK128;
    __bf16* Bp_out1 = Bp_root + PK128;
    __bf16* Bp_rel  = Bp_out1 + PK128;               // 5 consecutive

    const dim3 blk(256);
    const int gm  = (N + 31) / 32;
    const int gsc = (E * 32 + 255) / 256;

    // ---- pack all weight matrices into WMMA fragment layout ----
    pack_b<<<(int)((PK768 + 255) / 256), blk, 0, stream>>>(W_des,  Bp_des,  DDES);
    pack_b<<<(int)((PK128 + 255) / 256), blk, 0, stream>>>(W_in,   Bp_in,   DMODEL);
    pack_b<<<(int)((PK128 + 255) / 256), blk, 0, stream>>>(W_root, Bp_root, DMODEL);
    pack_b<<<(int)((PK128 + 255) / 256), blk, 0, stream>>>(W_out1, Bp_out1, DMODEL);
    for (int r = 0; r < NREL; ++r)
        pack_b<<<(int)((PK128 + 255) / 256), blk, 0, stream>>>(
            W_rel + (size_t)r * DMODEL * DMODEL, Bp_rel + (size_t)r * PK128, DMODEL);

    // x = leaky(des @ W_des + b_des); x = leaky(x @ W_in + b_in)
    gemm_nc128<1, 1><<<gm, blk, 0, stream>>>(des,  Bp_des, b_des, bufA, N, DDES);
    gemm_nc128<1, 1><<<gm, blk, 0, stream>>>(bufA, Bp_in,  b_in,  bufB, N, DMODEL);

    // per-(dst,relation) in-degree counts (shared by both RGCN layers)
    zerof<<<(N * NREL + 255) / 256, blk, 0, stream>>>(cnt, N * NREL);
    count_edges<<<(E + 255) / 256, blk, 0, stream>>>(dstp, etyp, cnt, E);

    // RGCN layer 1: bufB -> bufA
    gemm_nc128<0, 1><<<gm, blk, 0, stream>>>(bufB, Bp_root, b_rgcn, bufA, N, DMODEL);
    for (int r = 0; r < NREL; ++r) {
        gemm_nc128<0, 0><<<gm, blk, 0, stream>>>(bufB, Bp_rel + (size_t)r * PK128,
                                                 nullptr, Hbuf, N, DMODEL);
        scatter_rel<<<gsc, blk, 0, stream>>>(srcp, dstp, etyp, cnt, Hbuf, bufA, E, r);
    }

    // RGCN layer 2: bufA -> bufB
    gemm_nc128<0, 1><<<gm, blk, 0, stream>>>(bufA, Bp_root, b_rgcn, bufB, N, DMODEL);
    for (int r = 0; r < NREL; ++r) {
        gemm_nc128<0, 0><<<gm, blk, 0, stream>>>(bufA, Bp_rel + (size_t)r * PK128,
                                                 nullptr, Hbuf, N, DMODEL);
        scatter_rel<<<gsc, blk, 0, stream>>>(srcp, dstp, etyp, cnt, Hbuf, bufB, E, r);
    }

    // y = leaky(x @ W_out1 + b_out1); out = y @ W_out2 + b_out2
    gemm_nc128<1, 1><<<gm, blk, 0, stream>>>(bufB, Bp_out1, b_out1, bufA, N, DMODEL);
    out_proj2<<<(N + 255) / 256, blk, 0, stream>>>(bufA, W_out2, b_out2, (float*)d_out, N);
}
